// FastSelfAttention_61048665145526
// MI455X (gfx1250) — matrix operand order repaired and checked by reading, not compiled, not run on hardware
//
#include <hip/hip_runtime.h>
#include <math.h>

// Problem constants (from reference)
#define D_MODEL 1024
#define NHEAD   16
#define DK      64
#define MFEAT   256
#define BATCH   4
#define SEQ     4096
#define NH      (BATCH * NHEAD)   // 64 batch-head pairs
#define NROWS   (BATCH * SEQ)     // 16384 token rows
#define EPSF    1e-6f

typedef unsigned short u16;
typedef unsigned int   u32;
typedef __attribute__((ext_vector_type(16))) __bf16 v16bf;
typedef __attribute__((ext_vector_type(8)))  float  v8f;

// ---------- bf16 helpers (round-to-nearest-even) ----------
__device__ __forceinline__ u16 f2bf(float f) {
  union { float f; u32 u; } v; v.f = f;
  u32 r = v.u + 0x7FFFu + ((v.u >> 16) & 1u);
  return (u16)(r >> 16);
}
__device__ __forceinline__ float bf2f(u16 h) {
  union { u32 u; float f; } v; v.u = ((u32)h) << 16;
  return v.f;
}

union FragCast { uint4 q[2]; v16bf v; };

// ---------- WMMA fragment loaders: all contiguous (2x ds_load_b128 each) ----------
// A-matrix 16x32 bf16 layout: lane&15 = row M, element j -> K = 16*(j>>3) + 8*hi + (j&7).
__device__ __forceinline__ v16bf load_frag_a_rm(const u16* sm, int row0, int k0,
                                                int ld, int lane) {
  const int r  = row0 + (lane & 15);
  const int hi = (lane >> 4) & 1;
  const u16* p = sm + r * ld + k0 + hi * 8;
  FragCast u;
  u.q[0] = *reinterpret_cast<const uint4*>(p);       // K = base+0..7
  u.q[1] = *reinterpret_cast<const uint4*>(p + 16);  // K = base+16..23
  return u.v;
}
// B-matrix 32x16 bf16 layout: lane&15 = column N, element j -> K = 16*(j>>3) + 8*hi + (j&7).
__device__ __forceinline__ v16bf load_frag_b_cm(const u16* sm, int col0, int k0,
                                                int ldk, int lane) {
  const int c  = col0 + (lane & 15);
  const int hi = (lane >> 4) & 1;
  const u16* p = sm + c * ldk + k0 + hi * 8;
  FragCast u;
  u.q[0] = *reinterpret_cast<const uint4*>(p);
  u.q[1] = *reinterpret_cast<const uint4*>(p + 16);
  return u.v;
}

__device__ __forceinline__ v8f wmma_bf16(v16bf a, v16bf b, v8f c) {
  return __builtin_amdgcn_wmma_f32_16x16x32_bf16(false, a, false, b,
                                                 (short)0, c, false, false);
}

// ---------- conversion kernels ----------
__global__ void cvt_f32_to_bf16_kernel(const float* __restrict__ src,
                                       u16* __restrict__ dst, int count) {
  int i = blockIdx.x * blockDim.x + threadIdx.x;
  if (i < count) dst[i] = f2bf(src[i]);
}
// dst (C x R) bf16 = transpose of src (R x C) fp32
__global__ void transpose_to_bf16_kernel(const float* __restrict__ src,
                                         u16* __restrict__ dst, int R, int C) {
  int i = blockIdx.x * blockDim.x + threadIdx.x;
  if (i >= R * C) return;
  const int r = i / C, c = i % C;
  dst[(size_t)c * R + r] = f2bf(src[(size_t)r * C + c]);
}

// ---------- generic WMMA GEMM, 64x128 block tile, 2x2 wave tile ----------
// 8 waves as 2(M) x 4(N); each wave: 2 A-frags + 2 B-frags -> 4 WMMAs per K-step.
// MODE 0: head-split bf16 out; MODE 1: fp32 row-major out.
template <int MODE>
__global__ void wmma_gemm_kernel(const u16* __restrict__ A,
                                 const u16* __restrict__ Bm,
                                 const float* __restrict__ bias,
                                 void* __restrict__ Cout,
                                 int M, int N, int K) {
  __shared__ u16 sA[64 * 40];    // [row][k]
  __shared__ u16 sB[128 * 40];   // [col][k]  (column-major swizzle)
  const int t    = threadIdx.x;
  const int lane = t & 31;
  const int w    = t >> 5;
  const int wmi  = w & 1;   // 32-row slab
  const int wni  = w >> 1;  // 32-col slab
  const int m0   = blockIdx.y * 64;
  const int n0   = blockIdx.x * 128;

  const int ra = t >> 2;            // 0..63   A tile row
  const int ca = (t & 3) * 8;       // 0..24   A tile k chunk
  const int rb = t >> 3;            // 0..31   B tile k row
  const int cb = (t & 7) * 16;      // 0..112  B tile col chunk

  v8f acc00 = {}, acc01 = {}, acc10 = {}, acc11 = {};
  for (int k0 = 0; k0 < K; k0 += 32) {
    // A: row-major 128-bit copy
    *reinterpret_cast<uint4*>(&sA[ra * 40 + ca]) =
        *reinterpret_cast<const uint4*>(&A[(size_t)(m0 + ra) * K + k0 + ca]);
    // B: read 16 contiguous columns of one k-row, scatter into column-major LDS
    {
      uint4 v0 = *reinterpret_cast<const uint4*>(&Bm[(size_t)(k0 + rb) * N + n0 + cb]);
      uint4 v1 = *reinterpret_cast<const uint4*>(&Bm[(size_t)(k0 + rb) * N + n0 + cb + 8]);
      const u16* e0 = reinterpret_cast<const u16*>(&v0);
      const u16* e1 = reinterpret_cast<const u16*>(&v1);
#pragma unroll
      for (int j = 0; j < 8; ++j) {
        sB[(cb + j) * 40 + rb]     = e0[j];
        sB[(cb + 8 + j) * 40 + rb] = e1[j];
      }
    }
    if (k0 + 32 < K) {
      __builtin_prefetch(&A[(size_t)(m0 + ra) * K + k0 + 32 + ca], 0, 0);
      __builtin_prefetch(&Bm[(size_t)(k0 + 32 + rb) * N + n0 + cb], 0, 0);
    }
    __syncthreads();
    v16bf a0 = load_frag_a_rm(sA, wmi * 32, 0, 40, lane);
    v16bf a1 = load_frag_a_rm(sA, wmi * 32 + 16, 0, 40, lane);
    v16bf b0 = load_frag_b_cm(sB, wni * 32, 0, 40, lane);
    v16bf b1 = load_frag_b_cm(sB, wni * 32 + 16, 0, 40, lane);
    acc00 = wmma_bf16(a0, b0, acc00);
    acc01 = wmma_bf16(a0, b1, acc01);
    acc10 = wmma_bf16(a1, b0, acc10);
    acc11 = wmma_bf16(a1, b1, acc11);
    __syncthreads();
  }

  const int hi = (lane >> 4) & 1;
#pragma unroll
  for (int mi = 0; mi < 2; ++mi) {
#pragma unroll
    for (int ni = 0; ni < 2; ++ni) {
      const v8f acc = (mi == 0) ? (ni == 0 ? acc00 : acc01)
                                : (ni == 0 ? acc10 : acc11);
      const int gc = n0 + wni * 32 + ni * 16 + (lane & 15);
      const float bb = bias[gc];
#pragma unroll
      for (int i = 0; i < 8; ++i) {
        const int gr = m0 + wmi * 32 + mi * 16 + i + hi * 8;
        const float v = acc[i] + bb;
        if (MODE == 0) {
          u16* outp = (u16*)Cout;
          const int b = gr >> 12, l = gr & (SEQ - 1);
          const int h = gc >> 6, d = gc & 63;
          outp[(((size_t)(b * NHEAD + h) * SEQ + l) << 6) + d] = f2bf(v);
        } else {
          ((float*)Cout)[(size_t)gr * N + gc] = v;
        }
      }
    }
  }
}

// ---------- FAVOR+ feature map: Phi = (exp(q @ orfT * s1 - |q|^2/16) + eps)/16 ----------
__global__ void feature_map_kernel(const u16* __restrict__ Qh,   // [NH][SEQ][64]
                                   const u16* __restrict__ OrfT, // [64][256]
                                   u16* __restrict__ Phi) {      // [NH][SEQ][256]
  __shared__ u16  sQ[32 * 72];   // [l][k] row-major
  __shared__ u16  sO[64 * 72];   // [mcol][k] column-major swizzle
  __shared__ float sNorm[32];
  const int t     = threadIdx.x;
  const int lane  = t & 31;
  const int w     = t >> 5;
  const int wl    = w >> 2;  // L sub tile (0..1)
  const int wmc   = w & 3;   // feature sub tile (0..3)
  const int mbase = blockIdx.x * 64;
  const int l0    = blockIdx.y * 32;
  const int n     = blockIdx.z;

  if (t < 32) sNorm[t] = 0.f;
  __syncthreads();
  {
    const int r = t >> 3;        // 0..31 (l)
    const int c = (t & 7) * 8;   // 0..56 (k)
    uint4 v = *reinterpret_cast<const uint4*>(
        &Qh[((size_t)n * SEQ + l0 + r) * 64 + c]);
    *reinterpret_cast<uint4*>(&sQ[r * 72 + c]) = v;
    const u16* e = reinterpret_cast<const u16*>(&v);
    float ss = 0.f;
#pragma unroll
    for (int j = 0; j < 8; ++j) { float f = bf2f(e[j]); ss += f * f; }
    atomicAdd(&sNorm[r], ss);    // ds_add_f32
  }
  {
    const int r = t >> 2;        // 0..63 (k row of orfT)
    const int c = (t & 3) * 16;  // 0..48 (m col chunk)
    uint4 v0 = *reinterpret_cast<const uint4*>(&OrfT[(size_t)r * MFEAT + mbase + c]);
    uint4 v1 = *reinterpret_cast<const uint4*>(&OrfT[(size_t)r * MFEAT + mbase + c + 8]);
    const u16* e0 = reinterpret_cast<const u16*>(&v0);
    const u16* e1 = reinterpret_cast<const u16*>(&v1);
#pragma unroll
    for (int j = 0; j < 8; ++j) {
      sO[(c + j) * 72 + r]     = e0[j];
      sO[(c + 8 + j) * 72 + r] = e1[j];
    }
  }
  __syncthreads();

  v8f acc = {};
#pragma unroll
  for (int k0 = 0; k0 < 64; k0 += 32) {
    v16bf a = load_frag_a_rm(sQ, wl * 16, k0, 72, lane);
    v16bf b = load_frag_b_cm(sO, wmc * 16, k0, 72, lane);
    acc = wmma_bf16(a, b, acc);
  }

  const float s1 = 0.35355339059327373f;  // 64^-0.25
  const int hi = (lane >> 4) & 1;
#pragma unroll
  for (int i = 0; i < 8; ++i) {
    const int lr  = wl * 16 + i + hi * 8;
    const int mc  = mbase + wmc * 16 + (lane & 15);
    const float nrm = sNorm[lr] * 0.0625f;          // /(2*sqrt(64))
    const float p = (__expf(acc[i] * s1 - nrm) + EPSF) * 0.0625f;  // /sqrt(256)
    Phi[((size_t)n * SEQ + l0 + lr) * MFEAT + mc] = f2bf(p);
  }
}

// ---------- kv = Phi_k^T (256xL) @ V (Lx64), 128x64 block, 2x2 wave tile ----------
__global__ void kv_accum_kernel(const u16* __restrict__ PhiK, // [NH][SEQ][256]
                                const u16* __restrict__ Vh,   // [NH][SEQ][64]
                                float* __restrict__ KV) {     // [NH][256][64]
  __shared__ u16 sKt[128 * 40]; // [m][l]  (transposed at store time)
  __shared__ u16 sV[64 * 40];   // [d][l]  (column-major swizzle)
  const int t     = threadIdx.x;
  const int lane  = t & 31;
  const int w     = t >> 5;
  const int wmi   = w & 3;   // 32-m slab
  const int wdi   = w >> 2;  // 32-d slab
  const int mbase = blockIdx.x * 128;
  const int n     = blockIdx.y;

  const int r  = t >> 3;        // 0..31 (l row)
  const int ck = (t & 7) * 16;  // 0..112 PhiK col chunk
  const int cv = (t & 7) * 8;   // 0..56  V col chunk

  v8f acc00 = {}, acc01 = {}, acc10 = {}, acc11 = {};
  for (int l0 = 0; l0 < SEQ; l0 += 32) {
    {
      uint4 k0v = *reinterpret_cast<const uint4*>(
          &PhiK[((size_t)n * SEQ + l0 + r) * MFEAT + mbase + ck]);
      uint4 k1v = *reinterpret_cast<const uint4*>(
          &PhiK[((size_t)n * SEQ + l0 + r) * MFEAT + mbase + ck + 8]);
      uint4 vv = *reinterpret_cast<const uint4*>(
          &Vh[((size_t)n * SEQ + l0 + r) * 64 + cv]);
      const u16* ek0 = reinterpret_cast<const u16*>(&k0v);
      const u16* ek1 = reinterpret_cast<const u16*>(&k1v);
      const u16* ev  = reinterpret_cast<const u16*>(&vv);
#pragma unroll
      for (int j = 0; j < 8; ++j) {
        sKt[(ck + j) * 40 + r]     = ek0[j];
        sKt[(ck + 8 + j) * 40 + r] = ek1[j];
        sV[(cv + j) * 40 + r]      = ev[j];
      }
    }
    if (l0 + 32 < SEQ) {
      __builtin_prefetch(&PhiK[((size_t)n * SEQ + l0 + 32 + r) * MFEAT + mbase + ck], 0, 0);
      __builtin_prefetch(&Vh[((size_t)n * SEQ + l0 + 32 + r) * 64 + cv], 0, 0);
    }
    __syncthreads();
    v16bf a0 = load_frag_a_rm(sKt, wmi * 32, 0, 40, lane);     // A = Phi_k^T
    v16bf a1 = load_frag_a_rm(sKt, wmi * 32 + 16, 0, 40, lane);
    v16bf b0 = load_frag_b_cm(sV, wdi * 32, 0, 40, lane);
    v16bf b1 = load_frag_b_cm(sV, wdi * 32 + 16, 0, 40, lane);
    acc00 = wmma_bf16(a0, b0, acc00);
    acc01 = wmma_bf16(a0, b1, acc01);
    acc10 = wmma_bf16(a1, b0, acc10);
    acc11 = wmma_bf16(a1, b1, acc11);
    __syncthreads();
  }
  const int hi = (lane >> 4) & 1;
#pragma unroll
  for (int mi = 0; mi < 2; ++mi) {
#pragma unroll
    for (int ni = 0; ni < 2; ++ni) {
      const v8f acc = (mi == 0) ? (ni == 0 ? acc00 : acc01)
                                : (ni == 0 ? acc10 : acc11);
      const int d0 = wdi * 32 + ni * 16 + (lane & 15);
#pragma unroll
      for (int i = 0; i < 8; ++i) {
        const int mr = mbase + wmi * 32 + mi * 16 + i + hi * 8;
        KV[((size_t)n * MFEAT + mr) * 64 + d0] = acc[i];
      }
    }
  }
}

// ---------- ksum = sum over L of Phi_k ----------
__global__ void zero_f32_kernel(float* p, int count) {
  int i = blockIdx.x * blockDim.x + threadIdx.x;
  if (i < count) p[i] = 0.f;
}
__global__ void ksum_kernel(const u16* __restrict__ PhiK, float* __restrict__ Ksum) {
  const int n = blockIdx.x;
  const int chunk = blockIdx.y;  // 512 L-rows each
  const int m = threadIdx.x;     // 0..255 (coalesced over feature dim)
  float acc = 0.f;
  const size_t base = ((size_t)n * SEQ + (size_t)chunk * 512) * MFEAT + m;
  for (int l = 0; l < 512; ++l) acc += bf2f(PhiK[base + (size_t)l * MFEAT]);
  atomicAdd(&Ksum[n * MFEAT + m], acc);
}

// ---------- BufExt[n] = [kv | ksum | 0...]  (256 x 80) bf16 ----------
__global__ void pack_buf_kernel(const float* __restrict__ KV,
                                const float* __restrict__ Ksum,
                                u16* __restrict__ BufExt) {
  int idx = blockIdx.x * blockDim.x + threadIdx.x;
  const int total = NH * MFEAT * 80;
  if (idx >= total) return;
  const int c = idx % 80;
  const int m = (idx / 80) % MFEAT;
  const int n = idx / (80 * MFEAT);
  float v;
  if (c < 64)       v = KV[((size_t)n * MFEAT + m) * 64 + c];
  else if (c == 64) v = Ksum[n * MFEAT + m];
  else              v = 0.f;
  BufExt[idx] = f2bf(v);
}

// ---------- out_tile = Phi_q @ BufExt, normalize, merge heads ----------
__global__ void attn_out_kernel(const u16* __restrict__ PhiQ,   // [NH][SEQ][256]
                                const u16* __restrict__ BufExt, // [NH][256][80]
                                u16* __restrict__ Tmp) {        // [B][SEQ][D] bf16
  __shared__ u16  sA[16 * 264];  // [l][k] row-major
  __shared__ u16  sB[80 * 264];  // [col][k] column-major swizzle
  __shared__ float sNum[16 * 80];
  const int t    = threadIdx.x;
  const int lane = t & 31;
  const int w    = t >> 5;
  const int l0   = blockIdx.x * 16;
  const int n    = blockIdx.y;

  {
    const int r = t >> 4;         // 0..15
    const int c = (t & 15) * 16;  // 0..240
    *reinterpret_cast<uint4*>(&sA[r * 264 + c]) =
        *reinterpret_cast<const uint4*>(&PhiQ[((size_t)n * SEQ + l0 + r) * MFEAT + c]);
    *reinterpret_cast<uint4*>(&sA[r * 264 + c + 8]) =
        *reinterpret_cast<const uint4*>(&PhiQ[((size_t)n * SEQ + l0 + r) * MFEAT + c + 8]);
  }
  {
    // BufExt row = 80 elems = 10 uint4; scatter into column-major LDS
    const uint4* src = reinterpret_cast<const uint4*>(BufExt + (size_t)n * (MFEAT * 80));
#pragma unroll
    for (int i = 0; i < 10; ++i) {
      const int u  = i * 256 + t;     // uint4 index, 0..2559
      const int k  = u / 10;          // 0..255
      const int c0 = (u % 10) * 8;    // 0..72
      uint4 v = src[u];
      const u16* e = reinterpret_cast<const u16*>(&v);
#pragma unroll
      for (int j = 0; j < 8; ++j) sB[(c0 + j) * 264 + k] = e[j];
    }
  }
  __syncthreads();

  if (w < 5) {  // 5 N-tiles of 16 cover the 80-wide buffer
    v8f acc = {};
#pragma unroll
    for (int k0 = 0; k0 < MFEAT; k0 += 32) {
      v16bf a = load_frag_a_rm(sA, 0, k0, 264, lane);
      v16bf b = load_frag_b_cm(sB, w * 16, k0, 264, lane);
      acc = wmma_bf16(a, b, acc);
    }
    const int hi = (lane >> 4) & 1;
#pragma unroll
    for (int i = 0; i < 8; ++i)
      sNum[(i + hi * 8) * 80 + w * 16 + (lane & 15)] = acc[i];
  }
  __syncthreads();

  const int b = n >> 4;
  const int h = n & 15;
#pragma unroll
  for (int j = 0; j < 4; ++j) {
    const int idx = j * 256 + t;  // 0..1023 = 16 rows x 64 cols
    const int r = idx >> 6;
    const int d = idx & 63;
    const float den = sNum[r * 80 + 64];
    const float val = sNum[r * 80 + d] / den;
    Tmp[(((size_t)b * SEQ + l0 + r) * D_MODEL) + h * 64 + d] = f2bf(val);
  }
}

// ---------- host orchestration ----------
extern "C" void kernel_launch(void* const* d_in, const int* in_sizes, int n_in,
                              void* d_out, int out_size, void* d_ws, size_t ws_size,
                              hipStream_t stream) {
  (void)in_sizes; (void)n_in; (void)out_size; (void)ws_size;
  const float* x   = (const float*)d_in[0];
  const float* Wq  = (const float*)d_in[1];
  const float* bq  = (const float*)d_in[2];
  const float* Wk  = (const float*)d_in[3];
  const float* bk  = (const float*)d_in[4];
  const float* Wv  = (const float*)d_in[5];
  const float* bv  = (const float*)d_in[6];
  const float* Wo  = (const float*)d_in[7];
  const float* bo  = (const float*)d_in[8];
  const float* orf = (const float*)d_in[9];
  float* out = (float*)d_out;

  char* ws = (char*)d_ws;
  size_t off = 0;
  auto alloc = [&](size_t bytes) -> void* {
    void* p = ws + off;
    off = (off + bytes + 255) & ~(size_t)255;
    return p;
  };
  u16* Xbf    = (u16*)alloc((size_t)NROWS * D_MODEL * 2);
  u16* WTq    = (u16*)alloc((size_t)D_MODEL * D_MODEL * 2);
  u16* WTk    = (u16*)alloc((size_t)D_MODEL * D_MODEL * 2);
  u16* WTv    = (u16*)alloc((size_t)D_MODEL * D_MODEL * 2);
  u16* WTo    = (u16*)alloc((size_t)D_MODEL * D_MODEL * 2);
  u16* OrfT   = (u16*)alloc((size_t)DK * MFEAT * 2);
  u16* Qh     = (u16*)alloc((size_t)NH * SEQ * DK * 2);
  u16* Kh     = (u16*)alloc((size_t)NH * SEQ * DK * 2);
  u16* Vh     = (u16*)alloc((size_t)NH * SEQ * DK * 2);
  u16* PhiQ   = (u16*)alloc((size_t)NH * SEQ * MFEAT * 2);
  u16* PhiK   = (u16*)alloc((size_t)NH * SEQ * MFEAT * 2);
  float* KV   = (float*)alloc((size_t)NH * MFEAT * DK * 4);
  float* Ksum = (float*)alloc((size_t)NH * MFEAT * 4);
  u16* BufExt = (u16*)alloc((size_t)NH * MFEAT * 80 * 2);
  u16* Tmp    = (u16*)alloc((size_t)NROWS * D_MODEL * 2);

  // 1) precision conversion
  {
    const int cnt = NROWS * D_MODEL;
    cvt_f32_to_bf16_kernel<<<(cnt + 255) / 256, 256, 0, stream>>>(x, Xbf, cnt);
  }
  {
    const int cnt = D_MODEL * D_MODEL;
    const int g = (cnt + 255) / 256;
    transpose_to_bf16_kernel<<<g, 256, 0, stream>>>(Wq, WTq, D_MODEL, D_MODEL);
    transpose_to_bf16_kernel<<<g, 256, 0, stream>>>(Wk, WTk, D_MODEL, D_MODEL);
    transpose_to_bf16_kernel<<<g, 256, 0, stream>>>(Wv, WTv, D_MODEL, D_MODEL);
    transpose_to_bf16_kernel<<<g, 256, 0, stream>>>(Wo, WTo, D_MODEL, D_MODEL);
  }
  {
    const int cnt = MFEAT * DK;
    transpose_to_bf16_kernel<<<(cnt + 255) / 256, 256, 0, stream>>>(orf, OrfT, MFEAT, DK);
  }

  // 2) q/k/v projections (WMMA bf16, head-split bf16 epilogue)
  dim3 gemmGrid(D_MODEL / 128, NROWS / 64);   // 64x128 block tiles
  wmma_gemm_kernel<0><<<gemmGrid, 256, 0, stream>>>(Xbf, WTq, bq, Qh, NROWS, D_MODEL, D_MODEL);
  wmma_gemm_kernel<0><<<gemmGrid, 256, 0, stream>>>(Xbf, WTk, bk, Kh, NROWS, D_MODEL, D_MODEL);
  wmma_gemm_kernel<0><<<gemmGrid, 256, 0, stream>>>(Xbf, WTv, bv, Vh, NROWS, D_MODEL, D_MODEL);

  // 3) FAVOR+ feature maps
  dim3 featGrid(MFEAT / 64, SEQ / 32, NH);
  feature_map_kernel<<<featGrid, 256, 0, stream>>>(Qh, OrfT, PhiQ);
  feature_map_kernel<<<featGrid, 256, 0, stream>>>(Kh, OrfT, PhiK);

  // 4) kv and ksum
  dim3 kvGrid(MFEAT / 128, NH);               // 128x64 block tiles
  kv_accum_kernel<<<kvGrid, 256, 0, stream>>>(PhiK, Vh, KV);
  zero_f32_kernel<<<(NH * MFEAT + 255) / 256, 256, 0, stream>>>(Ksum, NH * MFEAT);
  dim3 ksGrid(NH, SEQ / 512);
  ksum_kernel<<<ksGrid, 256, 0, stream>>>(PhiK, Ksum);
  pack_buf_kernel<<<(NH * MFEAT * 80 + 255) / 256, 256, 0, stream>>>(KV, Ksum, BufExt);

  // 5) attention output + normalization + head merge
  dim3 aoGrid(SEQ / 16, NH);
  attn_out_kernel<<<aoGrid, 256, 0, stream>>>(PhiQ, BufExt, Tmp);

  // 6) final projection @ Wo^T + bo -> fp32 output
  wmma_gemm_kernel<1><<<gemmGrid, 256, 0, stream>>>(Tmp, WTo, bo, out, NROWS, D_MODEL, D_MODEL);
}
